// MultiHeadAttention_7980049236264
// MI455X (gfx1250) — matrix operand order repaired and checked by reading, not compiled, not run on hardware
//
#include <hip/hip_runtime.h>

typedef __attribute__((ext_vector_type(16))) _Float16 v16h;
typedef __attribute__((ext_vector_type(8)))  _Float16 v8h;
typedef __attribute__((ext_vector_type(8)))  float    v8f;

#define WMMA_F16(A,B,C) __builtin_amdgcn_wmma_f32_16x16x32_f16( \
    false, (A), false, (B), (short)0, (C), false, false)

// Problem constants (from reference)
#define BB 8
#define SS 1024
#define EE 128
#define HH 8
#define DQ 128
#define HD 1024

// Load a v16h fragment as two 128-bit loads (global or LDS), `gap` halves apart.
static __device__ __forceinline__ v16h frag16(const _Float16* p, int gap)
{
    v8h lo = *(const v8h*)p;
    v8h hv = *(const v8h*)(p + gap);
    return __builtin_shufflevector(lo, hv, 0, 1, 2, 3, 4, 5, 6, 7,
                                           8, 9, 10, 11, 12, 13, 14, 15);
}

// ---------------------------------------------------------------------------
// Packed fragment-image layouts (element = per-lane register image):
//   Qp : [bh][st(64)][j(4)]  [lane(32)][half(16)]   A-frag of Q (16 s x 32 d)
//   Kp : [bh][kt(64)][j(4)]  [lane(32)][half(16)]   B-frag of K^T (32 d x 16 key)
//   Vp : [bh][c(32)] [j(8)]  [lane(32)][half(16)]   B-frag of V (32 key x 16 d)
//   AOp: [b][st(64)] [ko(32)][lane(32)][half(16)]   A-frag of attn out (16 s x 32 o)
// ---------------------------------------------------------------------------

// ---------------------------------------------------------------------------
// Kernel 1: Q/K/V projections.  One wave computes a 16x16 tile of x @ W^T via
// 4 WMMA k-steps over E=128, then scatters the D tile into fragment-image
// layout so the attention kernel reads pure b128 fragments.
// ---------------------------------------------------------------------------
__global__ void proj_kernel(const float* __restrict__ q, const float* __restrict__ k,
                            const float* __restrict__ v,
                            const float* __restrict__ Wq, const float* __restrict__ Wk,
                            const float* __restrict__ Wv,
                            _Float16* __restrict__ Qp, _Float16* __restrict__ Kp,
                            _Float16* __restrict__ Vp)
{
    const int wid  = blockIdx.x * (blockDim.x >> 5) + (threadIdx.x >> 5);
    const int lane = threadIdx.x & 31;
    const int ln   = lane & 15;
    const int hi   = lane >> 4;

    const int mid  = wid / 32768;          // 0=Q, 1=K, 2=V
    const int rem  = wid % 32768;
    const int b    = rem >> 12;
    const int rem2 = rem & 4095;
    const int s0   = (rem2 >> 6) * 16;     // s-tile base
    const int o0   = (rem2 & 63) * 16;     // o-tile base

    const float* x = (mid == 0) ? q  : (mid == 1) ? k  : v;
    const float* W = (mid == 0) ? Wq : (mid == 1) ? Wk : Wv;

    v8f acc = {};
    for (int e0 = 0; e0 < EE; e0 += 32) {
        v16h a, bf;
        const float* arow = x + ((size_t)b * SS + (s0 + ln)) * EE + e0;
#pragma unroll
        for (int h = 0; h < 16; ++h) {
            const int kk = h + (hi ? 16 : 0);
            a[h] = (_Float16)arow[kk];
        }
        const float* brow = W + (size_t)(o0 + ln) * EE + e0 + (hi ? 16 : 0);
#pragma unroll
        for (int h = 0; h < 16; ++h) bf[h] = (_Float16)brow[h];
        acc = WMMA_F16(a, bf, acc);
    }

    const int head  = o0 >> 7;
    const int dbase = o0 & 127;
    const size_t bh = (size_t)b * HH + head;
#pragma unroll
    for (int r = 0; r < 8; ++r) {
        const int m = r + 8 * hi;
        const int s = s0 + m;                  // seq row
        const int d = dbase + ln;              // d col within head
        const _Float16 val = (_Float16)acc[r];
        if (mid == 0) {                        // Q -> A-frag image
            const int j  = d >> 5, kd = d & 31;
            const int al = (s & 15) + ((kd & 8) ? 16 : 0);
            const int hh = (kd & 7) + ((kd & 16) ? 8 : 0);
            Qp[((((bh * 64 + (s >> 4)) * 4 + j) * 32 + al) << 4) + hh] = val;
        } else if (mid == 1) {                 // K -> B-frag image (d x key)
            const int j  = d >> 5, kd = d & 31;
            const int fl = (s & 15) + ((kd & 8) ? 16 : 0);
            const int hh = (kd & 7) + ((kd & 16) ? 8 : 0);
            Kp[((((bh * 64 + (s >> 4)) * 4 + j) * 32 + fl) << 4) + hh] = val;
        } else {                               // V -> B-frag image (key x d)
            const int c  = s >> 5, kk2 = s & 31;
            const int j  = d >> 4,  n  = d & 15;
            const int fl = n + ((kk2 & 16) ? 16 : 0);
            const int hh = kk2 & 15;
            Vp[((((bh * 32 + c) * 8 + j) * 32 + fl) << 4) + hh] = val;
        }
    }
}

// ---------------------------------------------------------------------------
// Kernel 2: flash attention.  One wave owns 16 query rows of one (b,h).
// All Q/K/V fragments are b128 loads of the packed images; P is relaid out
// C/D->A through LDS with b128 readback.
// ---------------------------------------------------------------------------
__global__ void attn_kernel(const _Float16* __restrict__ Qp, const _Float16* __restrict__ Kp,
                            const _Float16* __restrict__ Vp, const float* __restrict__ mask,
                            _Float16* __restrict__ AOp)
{
    __shared__ __align__(16) _Float16 plds[4][16][32];   // per-wave P staging

    const int warp = threadIdx.x >> 5;
    const int wid  = blockIdx.x * (blockDim.x >> 5) + warp;
    const int lane = threadIdx.x & 31;
    const int ln   = lane & 15;
    const int hi   = lane >> 4;

    const int b  = wid >> 9;
    const int h  = (wid >> 6) & 7;
    const int qt = wid & 63;
    const int s0 = qt * 16;
    const size_t bh = (size_t)b * HH + h;

    // resident Q A-fragments (4 d-chunks of 32)
    v16h aq[4];
#pragma unroll
    for (int j = 0; j < 4; ++j)
        aq[j] = frag16(Qp + ((((bh * 64 + qt) * 4 + j) * 32 + lane) << 4), 8);

    v8f O[8];
#pragma unroll
    for (int j = 0; j < 8; ++j) O[j] = (v8f){};
    float Mrow[8], Lrow[8];
#pragma unroll
    for (int r = 0; r < 8; ++r) { Mrow[r] = -3.0e38f; Lrow[r] = 0.0f; }
    const float scale = 0.0883883476483184f;   // 1/sqrt(128)

    for (int m0 = 0; m0 < SS; m0 += 32) {
        if (m0 + 32 < SS) {   // prefetch next chunk's K/V fragment regions
            __builtin_prefetch(Kp + ((size_t)(bh * 64 + (m0 >> 4) + 2) * 4 * 512)
                                  + (lane << 4), 0, 0);
            __builtin_prefetch(Vp + ((size_t)(bh * 32 + (m0 >> 5) + 1) * 8 * 512)
                                  + (lane << 4), 0, 0);
        }

        float p[2][8];
        float tmax[8];
#pragma unroll
        for (int r = 0; r < 8; ++r) tmax[r] = -3.0e38f;

        // scores: two 16-key sub-tiles
#pragma unroll
        for (int t = 0; t < 2; ++t) {
            const int kt = (m0 >> 4) + t;
            v8f sc = (v8f){};
#pragma unroll
            for (int j = 0; j < 4; ++j) {
                v16h bk = frag16(Kp + ((((bh * 64 + kt) * 4 + j) * 32 + lane) << 4), 8);
                sc = WMMA_F16(aq[j], bk, sc);
            }
            const float mval = mask[(size_t)b * SS + m0 + t * 16 + ln];
#pragma unroll
            for (int r = 0; r < 8; ++r) {
                const float xv = sc[r] * scale + mval;
                p[t][r] = xv;
                tmax[r] = fmaxf(tmax[r], xv);
            }
        }

        // 16-lane reductions (wave32: xor masks 1..8 stay within half)
#pragma unroll
        for (int r = 0; r < 8; ++r) {
            float xv = tmax[r];
            xv = fmaxf(xv, __shfl_xor(xv, 1));
            xv = fmaxf(xv, __shfl_xor(xv, 2));
            xv = fmaxf(xv, __shfl_xor(xv, 4));
            xv = fmaxf(xv, __shfl_xor(xv, 8));
            tmax[r] = xv;
        }

        float alpha[8];
#pragma unroll
        for (int r = 0; r < 8; ++r) {
            const float newM = fmaxf(Mrow[r], tmax[r]);
            alpha[r] = __expf(Mrow[r] - newM);
            Mrow[r]  = newM;
            float sacc = 0.0f;
#pragma unroll
            for (int t = 0; t < 2; ++t) {
                const float e = __expf(p[t][r] - newM);
                p[t][r] = e;
                sacc += e;
            }
            sacc += __shfl_xor(sacc, 1);
            sacc += __shfl_xor(sacc, 2);
            sacc += __shfl_xor(sacc, 4);
            sacc += __shfl_xor(sacc, 8);
            Lrow[r] = Lrow[r] * alpha[r] + sacc;
        }

#pragma unroll
        for (int j = 0; j < 8; ++j)
#pragma unroll
            for (int r = 0; r < 8; ++r) O[j][r] *= alpha[r];

        // P: C/D layout -> A layout through LDS
#pragma unroll
        for (int t = 0; t < 2; ++t)
#pragma unroll
            for (int r = 0; r < 8; ++r)
                plds[warp][r + 8 * hi][t * 16 + ln] = (_Float16)p[t][r];
        asm volatile("s_wait_dscnt 0" ::: "memory");

        // A-frag readback: lane<16 takes k 0-15, lane>=16 takes 16-31
        const v16h ap = frag16(&plds[warp][ln][hi * 16], 8);
        asm volatile("s_wait_dscnt 0" ::: "memory");

        // P @ V : 8 d-chunks of 16
        const int c = m0 >> 5;
#pragma unroll
        for (int j = 0; j < 8; ++j) {
            v16h bv = frag16(Vp + ((((bh * 32 + c) * 8 + j) * 32 + lane) << 4), 8);
            O[j] = WMMA_F16(ap, bv, O[j]);
        }
    }

    // normalize and scatter into AO A-fragment image for the output projection
#pragma unroll
    for (int j = 0; j < 8; ++j) {
        const int ko = h * 4 + (j >> 1);               // 32-wide k-chunk along HD
        const int kO = ((j & 1) << 4) + ln;            // k within chunk
        const int hh = kO & 15;
#pragma unroll
        for (int r = 0; r < 8; ++r) {
            const int m  = r + 8 * hi;
            const int al = m + ((kO & 16) ? 16 : 0);
            const float val = O[j][r] / (Lrow[r] + 1e-12f);
            AOp[((((size_t)b * 64 + qt) * 32 + ko) * 32 + al) * 16 + hh] = (_Float16)val;
        }
    }
}

// ---------------------------------------------------------------------------
// Kernel 3: output projection  out = attn @ Wc^T + bc.  A-frags are b128
// loads of the packed image; 32 WMMA k-steps over HD=1024.
// ---------------------------------------------------------------------------
__global__ void outproj_kernel(const _Float16* __restrict__ AOp, const float* __restrict__ Wc,
                               const float* __restrict__ bc, float* __restrict__ out)
{
    const int wid  = blockIdx.x * (blockDim.x >> 5) + (threadIdx.x >> 5);
    const int lane = threadIdx.x & 31;
    const int ln   = lane & 15;
    const int hi   = lane >> 4;

    const int b  = wid >> 9;
    const int st = (wid >> 3) & 63;
    const int e0 = (wid & 7) * 16;
    const int s0 = st * 16;

    v8f acc = {};
    for (int ko = 0; ko < 32; ++ko) {
        const v16h a = frag16(AOp + ((((size_t)b * 64 + st) * 32 + ko) * 32 + lane) * 16, 8);
        v16h bf;
        const float* brow = Wc + (size_t)(e0 + ln) * HD + ko * 32 + (hi ? 16 : 0);
#pragma unroll
        for (int h = 0; h < 16; ++h) bf[h] = (_Float16)brow[h];
        acc = WMMA_F16(a, bf, acc);
    }

#pragma unroll
    for (int r = 0; r < 8; ++r) {
        const int m = r + 8 * hi;
        out[((size_t)b * SS + s0 + m) * EE + e0 + ln] = acc[r] + bc[e0 + ln];
    }
}

// ---------------------------------------------------------------------------
extern "C" void kernel_launch(void* const* d_in, const int* in_sizes, int n_in,
                              void* d_out, int out_size, void* d_ws, size_t ws_size,
                              hipStream_t stream)
{
    (void)in_sizes; (void)n_in; (void)out_size; (void)ws_size;
    const float* q    = (const float*)d_in[0];
    const float* k    = (const float*)d_in[1];
    const float* v    = (const float*)d_in[2];
    const float* mask = (const float*)d_in[3];
    const float* Wq   = (const float*)d_in[4];
    const float* Wk   = (const float*)d_in[5];
    const float* Wv   = (const float*)d_in[6];
    const float* Wc   = (const float*)d_in[7];
    const float* bc   = (const float*)d_in[8];
    float* out = (float*)d_out;

    // workspace: Qp | Kp | Vp | AOp, each B*S*HD f16 = 16 MB (64 MB total)
    _Float16* Qp  = (_Float16*)d_ws;
    _Float16* Kp  = Qp + (size_t)BB * HH * SS * DQ;
    _Float16* Vp  = Kp + (size_t)BB * HH * SS * DQ;
    _Float16* AOp = Vp + (size_t)BB * HH * SS * DQ;

    proj_kernel<<<12288, 256, 0, stream>>>(q, k, v, Wq, Wk, Wv, Qp, Kp, Vp);
    attn_kernel<<<1024, 128, 0, stream>>>(Qp, Kp, Vp, mask, AOp);
    outproj_kernel<<<512, 256, 0, stream>>>(AOp, Wc, bc, out);
}
